// MSDAM_19791209299992
// MI455X (gfx1250) — matrix operand order repaired and checked by reading, not compiled, not run on hardware
//
#include <hip/hip_runtime.h>
#include <hip/hip_bf16.h>
#include <math.h>

typedef __attribute__((ext_vector_type(16))) _Float16 v16h;
typedef __attribute__((ext_vector_type(8)))  float    v8f;

__device__ __forceinline__ float gelu_exact(float x) {
    return 0.5f * x * (1.0f + erff(x * 0.70710678118654752f));
}

// ---------------- AvgPool3d (1,2,2)/(1,2,2), pad (0,1,1), count_include_pad ----------------
// 1024 planes of 64x64 -> 33x33
__global__ void avgpool_kernel(const float* __restrict__ in, float* __restrict__ out) {
    int idx = blockIdx.x * blockDim.x + threadIdx.x;
    if (idx >= 1024 * 33 * 33) return;
    int x  = idx % 33;
    int y  = (idx / 33) % 33;
    int pl = idx / (33 * 33);
    const float* p = in + (size_t)pl * 4096;
    int h0 = 2 * y - 1, w0 = 2 * x - 1;
    float s = 0.f;
#pragma unroll
    for (int dy = 0; dy < 2; ++dy) {
#pragma unroll
        for (int dx = 0; dx < 2; ++dx) {
            int hh = h0 + dy, ww = w0 + dx;
            int hc = min(max(hh, 0), 63);
            int wc = min(max(ww, 0), 63);
            float m = (hh >= 0 && hh < 64 && ww >= 0 && ww < 64) ? 1.f : 0.f;
            s += m * p[hc * 64 + wc];
        }
    }
    out[idx] = s * 0.25f;
}

// ---------------- grouped 3x3 offset conv via WMMA ----------------
// One wave = one 16(out-ch) x 16(pixels) tile; M tiles of {0..15},{16,17-pad} per group.
// Contraction index kappa = k*8 + c  (c = kappa&7, k = kappa>>3), K padded 72 -> 96.
template <int H, int W, int DIL, int NTILES>
__global__ void __launch_bounds__(256)
offconv_wmma_kernel(const float* __restrict__ x, const float* __restrict__ wt,
                    const float* __restrict__ bs, float* __restrict__ off) {
    constexpr int HW = H * W;
    int lane = threadIdx.x & 31;
    int half = lane >> 4;
    int col  = lane & 15;
    int wave = blockIdx.x * 8 + (threadIdx.x >> 5);
    int sp = wave % NTILES;
    int mt = (wave / NTILES) & 1;
    int g  = (wave / (NTILES * 2)) & 3;
    int n  = wave / (NTILES * 8);
    int base = sp * 16;

    // A fragments: weights, rows m = mt*16+col (valid rows m<18).
    int m = mt * 16 + col;
    float rowm = (m < 18) ? 1.f : 0.f;
    int occ = min(g * 18 + m, 71);
    v16h a[3];
#pragma unroll
    for (int q = 0; q < 3; ++q) {
#pragma unroll
        for (int i = 0; i < 16; ++i) {
            int kk = q * 32 + (i < 8 ? half * 8 + i : 16 + half * 8 + (i - 8));
            int c = kk & 7;
            int k = min(kk >> 3, 8);
            float msk = (kk < 72) ? rowm : 0.f;
            a[q][i] = (_Float16)(wt[occ * 72 + c * 9 + k] * msk);
        }
    }

    // B fragments: im2col taps for pixel p = base+col.
    int p  = base + col;
    int pc = min(p, HW - 1);
    int h  = pc / W, w = pc % W;
    float pm = (p < HW) ? 1.f : 0.f;
    v8f acc = {};
#pragma unroll
    for (int q = 0; q < 3; ++q) {
        v16h b;
#pragma unroll
        for (int i = 0; i < 16; ++i) {
            int kk = q * 32 + (i < 8 ? half * 8 + i : 16 + half * 8 + (i - 8));
            int c = kk & 7;
            int k = min(kk >> 3, 8);
            int iy = h + (k / 3 - 1) * DIL;
            int ix = w + (k % 3 - 1) * DIL;
            int iyc = min(max(iy, 0), H - 1);
            int ixc = min(max(ix, 0), W - 1);
            float msk = ((kk < 72) && (iy >= 0) && (iy < H) && (ix >= 0) && (ix < W)) ? pm : 0.f;
            const float* xp = x + (size_t)(n * 32 + g * 8 + c) * HW;
            b[i] = (_Float16)(xp[iyc * W + ixc] * msk);
        }
        acc = __builtin_amdgcn_wmma_f32_16x16x32_f16(false, a[q], false, b,
                                                     (short)0, acc, false, false);
    }

    if (p < HW) {
#pragma unroll
        for (int r = 0; r < 8; ++r) {
            int ml = mt * 16 + half * 8 + r;
            if (ml < 18) {
                int o = g * 18 + ml;
                off[(size_t)(n * 72 + o) * HW + p] = acc[r] + bs[o];
            }
        }
    }
}

// ---------------- deformable conv via WMMA: one wave = 16 out-ch x 16 pixels tile ----------------
template <int H, int W, int DIL, int NTILES>
__global__ void __launch_bounds__(256)
deform_wmma_kernel(const float* __restrict__ x, const float* __restrict__ off,
                   const float* __restrict__ wdef, const float* __restrict__ bdef,
                   float* __restrict__ out) {
    constexpr int HW = H * W;
    __shared__ int   s_y0[8][144];
    __shared__ int   s_x0[8][144];
    __shared__ float s_wy[8][144];
    __shared__ float s_wx[8][144];

    int wv   = threadIdx.x >> 5;
    int lane = threadIdx.x & 31;
    int half = lane >> 4;
    int col  = lane & 15;
    int wave = blockIdx.x * 8 + wv;
    int sp = wave % NTILES;
    int g  = (wave / NTILES) & 3;
    int n  = wave / (NTILES * 4);
    int base = sp * 16;

    // Stage 1: bilinear sample coords for (k, pixel), shared across the 8 input channels.
#pragma unroll
    for (int t = 0; t < 5; ++t) {
        int i = t * 32 + lane;
        if (i < 144) {
            int k = i >> 4, j = i & 15;
            int p = base + j;
            int y0i = -100000, x0i = -100000;
            float wy = 0.f, wx = 0.f;
            if (p < HW) {
                int h = p / W, w = p % W;
                const float* ob = off + (size_t)(n * 72 + (g * 9 + k) * 2) * HW + p;
                float dy = ob[0];
                float dx = ob[HW];
                float py = (float)h + (float)((k / 3 - 1) * DIL) + dy;
                float px = (float)w + (float)((k % 3 - 1) * DIL) + dx;
                float fy = floorf(py), fx = floorf(px);
                wy = py - fy; wx = px - fx;
                y0i = (int)fy; x0i = (int)fx;
            }
            s_y0[wv][i] = y0i; s_x0[wv][i] = x0i;
            s_wy[wv][i] = wy;  s_wx[wv][i] = wx;
        }
    }
    __syncthreads();

    // Stage 2: A fragments = group weights [16 x 72], kappa = k*8 + c, padded to K=96.
    v16h a[3];
#pragma unroll
    for (int q = 0; q < 3; ++q) {
#pragma unroll
        for (int i = 0; i < 16; ++i) {
            int kk = q * 32 + (i < 8 ? half * 8 + i : 16 + half * 8 + (i - 8));
            int c = kk & 7;
            int k = min(kk >> 3, 8);
            float msk = (kk < 72) ? 1.f : 0.f;
            a[q][i] = (_Float16)(wdef[(g * 16 + col) * 72 + c * 9 + k] * msk);
        }
    }

    // Stage 3: B = bilinear-sampled patches (32x16 f16 per tile) + WMMA accumulate.
    v8f acc = {};
#pragma unroll
    for (int q = 0; q < 3; ++q) {
        v16h b;
#pragma unroll
        for (int i = 0; i < 16; ++i) {
            int kk = q * 32 + (i < 8 ? half * 8 + i : 16 + half * 8 + (i - 8));
            int c = kk & 7;
            int k = min(kk >> 3, 8);
            float inb = (kk < 72) ? 1.f : 0.f;
            int si = k * 16 + col;
            int y0 = s_y0[wv][si], x0 = s_x0[wv][si];
            float wy = s_wy[wv][si], wx = s_wx[wv][si];
            int y1 = y0 + 1, x1 = x0 + 1;
            int yc0 = min(max(y0, 0), H - 1);
            int yc1 = min(max(y1, 0), H - 1);
            int xc0 = min(max(x0, 0), W - 1);
            int xc1 = min(max(x1, 0), W - 1);
            float my0 = (y0 >= 0 && y0 < H) ? 1.f : 0.f;
            float my1 = (y1 >= 0 && y1 < H) ? 1.f : 0.f;
            float mx0 = (x0 >= 0 && x0 < W) ? 1.f : 0.f;
            float mx1 = (x1 >= 0 && x1 < W) ? 1.f : 0.f;
            const float* xp = x + (size_t)(n * 32 + g * 8 + c) * HW;
            float v00 = xp[yc0 * W + xc0] * my0 * mx0;
            float v01 = xp[yc0 * W + xc1] * my0 * mx1;
            float v10 = xp[yc1 * W + xc0] * my1 * mx0;
            float v11 = xp[yc1 * W + xc1] * my1 * mx1;
            float v = (v00 * (1.f - wy) * (1.f - wx) + v01 * (1.f - wy) * wx
                     + v10 * wy * (1.f - wx)         + v11 * wy * wx) * inb;
            b[i] = (_Float16)v;
        }
        acc = __builtin_amdgcn_wmma_f32_16x16x32_f16(false, a[q], false, b,
                                                     (short)0, acc, false, false);
    }

    // Stage 4: epilogue — add bias, store pre-norm output.
    int p = base + col;
    if (p < HW) {
#pragma unroll
        for (int r = 0; r < 8; ++r) {
            int co = g * 16 + half * 8 + r;
            out[(size_t)(n * 64 + co) * HW + p] = acc[r] + bdef[co];
        }
    }
}

// ---------------- fused instance-norm + exact GELU: one block per (n, co) ----------------
template <int HW>
__global__ void innorm_gelu_kernel(float* __restrict__ buf) {
    __shared__ float ssum[256], ssq[256];
    __shared__ float smean, srstd;
    int nc = blockIdx.x;
    float* p = buf + (size_t)nc * HW;
    float s = 0.f, sq = 0.f;
    for (int i = threadIdx.x; i < HW; i += 256) {
        float v = p[i]; s += v; sq += v * v;
    }
    ssum[threadIdx.x] = s; ssq[threadIdx.x] = sq;
    __syncthreads();
#pragma unroll
    for (int st = 128; st > 0; st >>= 1) {
        if ((int)threadIdx.x < st) {
            ssum[threadIdx.x] += ssum[threadIdx.x + st];
            ssq[threadIdx.x]  += ssq[threadIdx.x + st];
        }
        __syncthreads();
    }
    if (threadIdx.x == 0) {
        float m   = ssum[0] / (float)HW;
        float var = ssq[0] / (float)HW - m * m;
        smean = m;
        srstd = rsqrtf(var + 1e-5f);
    }
    __syncthreads();
    float m = smean, r = srstd;
    for (int i = threadIdx.x; i < HW; i += 256) {
        p[i] = gelu_exact((p[i] - m) * r);
    }
}

// ---------------- cosine similarity over 1024 channels (scale 1, HW=4096) ----------------
__global__ void cos_direct_kernel(const float* __restrict__ hb, const float* __restrict__ lb,
                                  float* __restrict__ sim) {
    int idx = blockIdx.x * blockDim.x + threadIdx.x;
    if (idx >= 2 * 4096) return;
    int p = idx & 4095;
    int b = idx >> 12;
    float num = 0.f, nh = 0.f, nl = 0.f;
    for (int d = 0; d < 16; ++d) {
        int n = b * 16 + d;
        const float* hp = hb + (size_t)n * 64 * 4096 + p;
        const float* lp = lb + (size_t)n * 64 * 4096 + p;
        for (int co = 0; co < 64; ++co) {
            float hv = hp[(size_t)co * 4096];
            float lv = lp[(size_t)co * 4096];
            num += hv * lv; nh += hv * hv; nl += lv * lv;
        }
    }
    sim[idx] = num / (fmaxf(sqrtf(nh), 1e-8f) * fmaxf(sqrtf(nl), 1e-8f));
}

// ---------------- cosine sim with inline 33->64 adaptive upsample (scale 0) ----------------
__global__ void cos_upsample_kernel(const float* __restrict__ hb, const float* __restrict__ lb,
                                    float* __restrict__ sim) {
    int idx = blockIdx.x * blockDim.x + threadIdx.x;   // 2*64*64
    if (idx >= 2 * 64 * 64) return;
    int xq = idx & 63;
    int yq = (idx >> 6) & 63;
    int b  = idx >> 12;
    int sy = (yq * 33) >> 6;
    int ey = ((yq + 1) * 33 + 63) >> 6;
    int sx = (xq * 33) >> 6;
    int ex = ((xq + 1) * 33 + 63) >> 6;
    float wgt = (1.f / (float)(ey - sy)) * (1.f / (float)(ex - sx));
    float num = 0.f, nh = 0.f, nl = 0.f;
    for (int d = 0; d < 16; ++d) {
        int n = b * 16 + d;
        for (int co = 0; co < 64; ++co) {
            const float* hp = hb + (size_t)(n * 64 + co) * 1089;
            const float* lp = lb + (size_t)(n * 64 + co) * 1089;
            float hv = 0.f, lv = 0.f;
            for (int iy = sy; iy < ey; ++iy)
                for (int ix = sx; ix < ex; ++ix) {
                    hv += wgt * hp[iy * 33 + ix];
                    lv += wgt * lp[iy * 33 + ix];
                }
            num += hv * lv; nh += hv * hv; nl += lv * lv;
        }
    }
    sim[idx] = num / (fmaxf(sqrtf(nh), 1e-8f) * fmaxf(sqrtf(nl), 1e-8f));
}

// ---------------- 1x1 fuse conv + GELU ----------------
__global__ void fuse_kernel(const float* __restrict__ sim0, const float* __restrict__ sim1,
                            const float* __restrict__ wf, const float* __restrict__ bf,
                            float* __restrict__ out) {
    int idx = blockIdx.x * blockDim.x + threadIdx.x;   // 2*128*4096
    if (idx >= 2 * 128 * 4096) return;
    int p = idx & 4095;
    int o = (idx >> 12) & 127;
    int b = idx >> 19;
    float v = wf[o * 2 + 0] * sim0[b * 4096 + p]
            + wf[o * 2 + 1] * sim1[b * 4096 + p] + bf[o];
    out[idx] = gelu_exact(v);
}

extern "C" void kernel_launch(void* const* d_in, const int* in_sizes, int n_in,
                              void* d_out, int out_size, void* d_ws, size_t ws_size,
                              hipStream_t stream) {
    (void)in_sizes; (void)n_in; (void)out_size; (void)ws_size;
    const float* hsi    = (const float*)d_in[0];
    const float* lid    = (const float*)d_in[1];
    const float* w_off0 = (const float*)d_in[2];
    const float* b_off0 = (const float*)d_in[3];
    const float* w_def0 = (const float*)d_in[4];
    const float* b_def0 = (const float*)d_in[5];
    const float* w_off1 = (const float*)d_in[6];
    const float* b_off1 = (const float*)d_in[7];
    const float* w_def1 = (const float*)d_in[8];
    const float* b_def1 = (const float*)d_in[9];
    const float* w_fuse = (const float*)d_in[10];
    const float* b_fuse = (const float*)d_in[11];
    float* out = (float*)d_out;

    float* ws   = (float*)d_ws;
    float* pool = ws;                       // 1,115,136  (32*32*33*33)
    float* offb = pool + 1115136;           // 9,437,184  (32*72*64*64)
    float* hbuf = offb + 9437184;           // 8,388,608  (32*64*64*64)
    float* lbuf = hbuf + 8388608;           // 8,388,608
    float* sim0 = lbuf + 8388608;           // 8,192
    float* sim1 = sim0 + 8192;              // 8,192

    // ---- scale 0: avgpool to 33x33, dilation 1 (NTILES = ceil(1089/16) = 69) ----
    {
        constexpr int NT0 = 69;
        avgpool_kernel<<<(1024 * 33 * 33 + 255) / 256, 256, 0, stream>>>(hsi, pool);
        offconv_wmma_kernel<33, 33, 1, NT0><<<32 * 4 * 2 * NT0 / 8, 256, 0, stream>>>(
            pool, w_off0, b_off0, offb);
        deform_wmma_kernel<33, 33, 1, NT0><<<32 * 4 * NT0 / 8, 256, 0, stream>>>(
            pool, offb, w_def0, b_def0, hbuf);
        innorm_gelu_kernel<1089><<<2048, 256, 0, stream>>>(hbuf);

        avgpool_kernel<<<(1024 * 33 * 33 + 255) / 256, 256, 0, stream>>>(lid, pool);
        offconv_wmma_kernel<33, 33, 1, NT0><<<32 * 4 * 2 * NT0 / 8, 256, 0, stream>>>(
            pool, w_off0, b_off0, offb);
        deform_wmma_kernel<33, 33, 1, NT0><<<32 * 4 * NT0 / 8, 256, 0, stream>>>(
            pool, offb, w_def0, b_def0, lbuf);
        innorm_gelu_kernel<1089><<<2048, 256, 0, stream>>>(lbuf);

        cos_upsample_kernel<<<(8192 + 255) / 256, 256, 0, stream>>>(hbuf, lbuf, sim0);
    }
    // ---- scale 1: full resolution 64x64, dilation 2 (NTILES = 256) ----
    {
        constexpr int NT1 = 256;
        offconv_wmma_kernel<64, 64, 2, NT1><<<32 * 4 * 2 * NT1 / 8, 256, 0, stream>>>(
            hsi, w_off1, b_off1, offb);
        deform_wmma_kernel<64, 64, 2, NT1><<<32 * 4 * NT1 / 8, 256, 0, stream>>>(
            hsi, offb, w_def1, b_def1, hbuf);
        innorm_gelu_kernel<4096><<<2048, 256, 0, stream>>>(hbuf);

        offconv_wmma_kernel<64, 64, 2, NT1><<<32 * 4 * 2 * NT1 / 8, 256, 0, stream>>>(
            lid, w_off1, b_off1, offb);
        deform_wmma_kernel<64, 64, 2, NT1><<<32 * 4 * NT1 / 8, 256, 0, stream>>>(
            lid, offb, w_def1, b_def1, lbuf);
        innorm_gelu_kernel<4096><<<2048, 256, 0, stream>>>(lbuf);

        cos_direct_kernel<<<32, 256, 0, stream>>>(hbuf, lbuf, sim1);
    }
    fuse_kernel<<<4096, 256, 0, stream>>>(sim0, sim1, w_fuse, b_fuse, out);
}
